// Attention_3839700762874
// MI455X (gfx1250) — matrix operand order repaired and checked by reading, not compiled, not run on hardware
//
#include <hip/hip_runtime.h>

// ---------------------------------------------------------------------------
// Fused Luong bilinear attention for MI455X (gfx1250, wave32, WMMA).
//   wq = q @ w                   (fp32 VALU, ~2 GFLOP, negligible)
//   o  = softmax(wq @ k^T) @ v   (FlashAttention, f16 WMMA, ~69 GFLOP)
// scores never touch HBM (a 3-pass version would move 256 MB of fp32).
// Double-buffered LDS; K tiles staged with CDNA5 async global->LDS loads
// (ASYNCcnt), V tiles staged transposed so PV B-fragments are contiguous
// ds_load_b128. One workgroup barrier per 32-key block.
// ---------------------------------------------------------------------------

typedef _Float16 v16h __attribute__((ext_vector_type(16)));
typedef _Float16 v8h  __attribute__((ext_vector_type(8)));
typedef float    v8f  __attribute__((ext_vector_type(8)));

#define BATCH 4
#define LSEQ  4096
#define DIM   256
#define KB    32            // keys per inner iteration
#define NB    (LSEQ / KB)   // 128 key blocks
#define KROW  264           // padded halves per K-tile row  (528B: 16B aligned, bank stride 4)
#define VROW  40            // padded halves per V^T row     (80B: 16B aligned, bank stride 20)
#define PROW  40            // padded halves per P row

#define KLDSH (KB * KROW)           // 8448 halves
#define VLDSH (DIM * VROW)          // 10240 halves
#define BUFH  (KLDSH + VLDSH)       // 18688 halves per buffer
#define PBASE (2 * BUFH)            // P scratch after the two buffers
#define SMEM_HALVES (PBASE + 4 * 16 * PROW)
#define SMEM_BYTES  (SMEM_HALVES * 2)   // 79872 B dynamic LDS (gfx1250: up to 320KB/WG)

#define NEG_BIG (-3.4028234e38f)

// ---------------- kernel 1: wq = q @ w  (f32 -> f16) -----------------------
__global__ void wq_kernel(const float* __restrict__ q,
                          const float* __restrict__ w,
                          _Float16* __restrict__ wq) {
    __shared__ float qrow[DIM];
    const int row = blockIdx.x;            // b*LSEQ + qi
    const int t   = threadIdx.x;           // output column
    qrow[t] = q[(size_t)row * DIM + t];
    __syncthreads();
    float acc = 0.f;
#pragma unroll 8
    for (int i = 0; i < DIM; ++i) acc = fmaf(qrow[i], w[i * DIM + t], acc);
    wq[(size_t)row * DIM + t] = (_Float16)acc;
}

// ---------------- kernel 2: k,v fp32 -> f16 --------------------------------
__global__ void cvt_kernel(const float* __restrict__ k,
                           const float* __restrict__ v,
                           _Float16* __restrict__ kh,
                           _Float16* __restrict__ vh) {
    const size_t i = (size_t)blockIdx.x * blockDim.x + threadIdx.x;
    kh[i] = (_Float16)k[i];
    vh[i] = (_Float16)v[i];
}

// ---------------- kernel 3: flash attention --------------------------------
// Block = 128 threads = 4 waves. Each wave: 16 query rows. Block: 64 rows.
// Grid  = BATCH * (LSEQ/64) = 256 blocks.
__launch_bounds__(128, 1)
__global__ void flash_kernel(const _Float16* __restrict__ wq,
                             const _Float16* __restrict__ kh,
                             const _Float16* __restrict__ vh,
                             float* __restrict__ out) {
    extern __shared__ _Float16 smem[];     // [buf0 K|V^T][buf1 K|V^T][P x4 waves]

    const int tid  = threadIdx.x;
    const int wave = tid >> 5;
    const int lane = tid & 31;
    const int g    = lane >> 4;   // lane half-group
    const int ln   = lane & 15;

    const int blk = blockIdx.x;
    const int b   = blk / (LSEQ / 64);
    const int q0  = (blk % (LSEQ / 64)) * 64 + wave * 16;

    const _Float16* kbase = kh + (size_t)b * LSEQ * DIM;
    const _Float16* vbase = vh + (size_t)b * LSEQ * DIM;

    // ---- stage key-block kb into buffer `buf`: K async, V^T manual ----
    auto stage = [&](int kb, int buf) {
        _Float16* kl = smem + buf * BUFH;
        _Float16* vl = kl + KLDSH;
#pragma unroll
        for (int i = 0; i < 8; ++i) {           // 32 keys x 256 halves, 16B/lane/issue
            int s   = tid + 128 * i;
            int key = s >> 5;
            int off = (s & 31) * 8;
            const _Float16* gk = kbase + (size_t)(kb * KB + key) * DIM + off;
            unsigned int ldsoff = (unsigned int)(size_t)(kl + key * KROW + off);
            asm volatile("global_load_async_to_lds_b128 %0, %1, off"
                         :: "v"(ldsoff), "v"((unsigned long long)(size_t)gk)
                         : "memory");
        }
#pragma unroll
        for (int i = 0; i < 8; ++i) {           // V staged transposed: [dout][key]
            int s   = tid + 128 * i;
            int key = s >> 5;
            int off = (s & 31) * 8;
            v8h d = *(const v8h*)(vbase + (size_t)(kb * KB + key) * DIM + off);
#pragma unroll
            for (int j = 0; j < 8; ++j) vl[(off + j) * VROW + key] = d[j];
        }
        if (kb + 1 < NB)                         // L2 prefetch one further tile
            __builtin_prefetch(vbase + (size_t)((kb + 1) * KB) * DIM + tid * 16, 0, 0);
    };

    // ---- load wq A-fragments (ISA 16-bit A layout, 8 Dout chunks) ----
    v16h afrag[8];
    {
        const _Float16* base = wq + ((size_t)(b * LSEQ + q0 + ln)) * DIM;
        const int c0 = g * 8;
#pragma unroll
        for (int c = 0; c < 8; ++c) {
            v8h lo = *(const v8h*)(base + c * 32 + c0);
            v8h hi = *(const v8h*)(base + c * 32 + c0 + 16);
            v16h a;
#pragma unroll
            for (int j = 0; j < 8; ++j) { a[j] = lo[j]; a[8 + j] = hi[j]; }
            afrag[c] = a;
        }
    }

    v8f o_acc[16];
#pragma unroll
    for (int t = 0; t < 16; ++t)
#pragma unroll
        for (int r = 0; r < 8; ++r) o_acc[t][r] = 0.f;

    float m_i[8], l_i[8];
#pragma unroll
    for (int r = 0; r < 8; ++r) { m_i[r] = NEG_BIG; l_i[r] = 0.f; }

    stage(0, 0);

    for (int kb = 0; kb < NB; ++kb) {
        const int cur = kb & 1;
        // our async issues into `cur` (from last iteration / prologue) must land,
        // then the block barrier publishes the buffer to all 4 waves.
        asm volatile("s_wait_asynccnt 0" ::: "memory");
        __syncthreads();
        if (kb + 1 < NB) stage(kb + 1, cur ^ 1);   // overlap next tile with compute

        const _Float16* kl = smem + cur * BUFH;
        const _Float16* vl = kl + KLDSH;

        // ---- scores: s[16q x 32k] = wq_tile @ k_tile^T, contraction over Dout ----
        v8f s0, s1;
#pragma unroll
        for (int r = 0; r < 8; ++r) { s0[r] = 0.f; s1[r] = 0.f; }
#pragma unroll
        for (int c = 0; c < 8; ++c) {
            const _Float16* kr0 = kl + (ln)      * KROW + c * 32 + g * 8;
            const _Float16* kr1 = kl + (16 + ln) * KROW + c * 32 + g * 8;
            v8h b0lo = *(const v8h*)(kr0);
            v8h b0hi = *(const v8h*)(kr0 + 16);
            v8h b1lo = *(const v8h*)(kr1);
            v8h b1hi = *(const v8h*)(kr1 + 16);
            v16h b0, b1;
#pragma unroll
            for (int j = 0; j < 8; ++j) {
                b0[j] = b0lo[j]; b0[8 + j] = b0hi[j];
                b1[j] = b1lo[j]; b1[8 + j] = b1hi[j];
            }
            s0 = __builtin_amdgcn_wmma_f32_16x16x32_f16(false, afrag[c], false, b0,
                                                        (short)0, s0, false, false);
            s1 = __builtin_amdgcn_wmma_f32_16x16x32_f16(false, afrag[c], false, b1,
                                                        (short)0, s1, false, false);
        }

        // ---- online softmax (row reductions across 16-lane half-groups) ----
        float alpha[8], p0[8], p1[8];
#pragma unroll
        for (int r = 0; r < 8; ++r) {
            float mx = fmaxf(s0[r], s1[r]);
            mx = fmaxf(mx, __shfl_xor(mx, 1));
            mx = fmaxf(mx, __shfl_xor(mx, 2));
            mx = fmaxf(mx, __shfl_xor(mx, 4));
            mx = fmaxf(mx, __shfl_xor(mx, 8));
            float mnew = fmaxf(m_i[r], mx);
            alpha[r] = __expf(m_i[r] - mnew);
            p0[r] = __expf(s0[r] - mnew);
            p1[r] = __expf(s1[r] - mnew);
            float rs = p0[r] + p1[r];
            rs += __shfl_xor(rs, 1);
            rs += __shfl_xor(rs, 2);
            rs += __shfl_xor(rs, 4);
            rs += __shfl_xor(rs, 8);
            l_i[r] = l_i[r] * alpha[r] + rs;
            m_i[r] = mnew;
        }
#pragma unroll
        for (int t = 0; t < 16; ++t)
#pragma unroll
            for (int r = 0; r < 8; ++r) o_acc[t][r] *= alpha[r];

        // ---- P: C-layout -> A-layout via per-wave LDS round-trip ----
        _Float16* pw = smem + PBASE + wave * 16 * PROW;
#pragma unroll
        for (int r = 0; r < 8; ++r) {
            int row = r + 8 * g;
            pw[row * PROW + ln]      = (_Float16)p0[r];
            pw[row * PROW + 16 + ln] = (_Float16)p1[r];
        }
        asm volatile("s_wait_dscnt 0" ::: "memory");   // same-wave LDS RAW
        v16h pf;
        {
            const _Float16* pr = pw + ln * PROW;
            const int c0 = g * 8;
            v8h lo = *(const v8h*)(pr + c0);
            v8h hi = *(const v8h*)(pr + c0 + 16);
#pragma unroll
            for (int j = 0; j < 8; ++j) { pf[j] = lo[j]; pf[8 + j] = hi[j]; }
        }

        // ---- O += P @ V : contraction over 32 keys, 16 Dout tiles ----
#pragma unroll
        for (int t = 0; t < 16; ++t) {
            const _Float16* vr = vl + (t * 16 + ln) * VROW + g * 8;
            v8h vlo = *(const v8h*)(vr);
            v8h vhi = *(const v8h*)(vr + 16);
            v16h vf;
#pragma unroll
            for (int j = 0; j < 8; ++j) { vf[j] = vlo[j]; vf[8 + j] = vhi[j]; }
            o_acc[t] = __builtin_amdgcn_wmma_f32_16x16x32_f16(false, pf, false, vf,
                                                              (short)0, o_acc[t], false, false);
        }
        // no trailing barrier: next iteration's top barrier separates readers
        // of this buffer from its next overwrite (two iterations away).
    }

    // ---- epilogue: O /= l, store fp32 ----
    float inv[8];
#pragma unroll
    for (int r = 0; r < 8; ++r) inv[r] = 1.f / l_i[r];
    float* ob = out + ((size_t)(b * LSEQ + q0)) * DIM;
#pragma unroll
    for (int t = 0; t < 16; ++t)
#pragma unroll
        for (int r = 0; r < 8; ++r)
            ob[(size_t)(r + 8 * g) * DIM + t * 16 + ln] = o_acc[t][r] * inv[r];
}

// ---------------------------------------------------------------------------
extern "C" void kernel_launch(void* const* d_in, const int* in_sizes, int n_in,
                              void* d_out, int out_size, void* d_ws, size_t ws_size,
                              hipStream_t stream) {
    const float* q = (const float*)d_in[0];
    const float* v = (const float*)d_in[1];
    const float* k = (const float*)d_in[2];
    const float* w = (const float*)d_in[3];
    float* out = (float*)d_out;

    const size_t elems = (size_t)BATCH * LSEQ * DIM;   // 4,194,304
    _Float16* wq16 = (_Float16*)d_ws;
    _Float16* kh   = wq16 + elems;
    _Float16* vh   = kh + elems;                        // 24 MB of f16 workspace

    // opt in to >64KB dynamic LDS (gfx1250 WGs may use up to 320KB)
    (void)hipFuncSetAttribute((const void*)flash_kernel,
                              hipFuncAttributeMaxDynamicSharedMemorySize,
                              SMEM_BYTES);

    wq_kernel<<<BATCH * LSEQ, DIM, 0, stream>>>(q, w, wq16);
    cvt_kernel<<<(int)(elems / 256), 256, 0, stream>>>(k, v, kh, vh);
    flash_kernel<<<BATCH * (LSEQ / 64), 128, SMEM_BYTES, stream>>>(wq16, kh, vh, out);
}
